// CrossAttentionDecoder_63909113364767
// MI455X (gfx1250) — matrix operand order repaired and checked by reading, not compiled, not run on hardware
//
#include <hip/hip_runtime.h>
#include <math.h>

// Flip to 0 to fall back to synchronous b128 staging if the async mnemonic
// is rejected by the assembler.
#define ASYNC_STAGE 1

// ---------------------------------------------------------------------------
// CDNA5 (gfx1250) wave32 WMMA bf16 building blocks.
// All fragment loads are arranged to be contiguous 16B chunks per lane.
// ---------------------------------------------------------------------------
typedef __attribute__((ext_vector_type(16))) __bf16 bf16x16;
typedef __attribute__((ext_vector_type(8)))  __bf16 bf16x8;
typedef __attribute__((ext_vector_type(8)))  float  floatx8;
typedef __attribute__((address_space(3)))    __bf16 lds_bf16;

static __device__ __forceinline__ floatx8 wmma_bf16(bf16x16 a, bf16x16 b, floatx8 c) {
  return __builtin_amdgcn_wmma_f32_16x16x32_bf16(false, a, false, b, (short)0, c,
                                                 false, false);
}

static __device__ __forceinline__ bf16x16 combine8(bf16x8 lo, bf16x8 hi) {
  bf16x16 r;
#pragma unroll
  for (int e = 0; e < 8; ++e) { r[e] = lo[e]; r[e + 8] = hi[e]; }
  return r;
}

// A fragment: 16(M) x 32(K), source row-major [M][K] with leading dim ld.
// ISA 7.12.2: lane m=lane&15, halves at K = koff+0..7 and K = 16+koff+0..7,
// koff = 8*(lane>=16)  ->  two contiguous 16B loads per lane.
static __device__ __forceinline__ bf16x16 load_a_frag(const __bf16* base, int ld) {
  const int lane = threadIdx.x & 31;
  const __bf16* p = base + (lane & 15) * ld + ((lane >> 4) << 3);
  return combine8(*(const bf16x8*)p, *(const bf16x8*)(p + 16));
}

// B fragment: 32(K) x 16(N), source stored TRANSPOSED, i.e. [N][K] row-major
// with leading dim ld. Lane n=lane&15, halves at K = koff+0..7 and K = 16+koff+0..7,
//   koff = 8*(lane>=16)  ->  two contiguous 16B loads per lane.
static __device__ __forceinline__ bf16x16 load_b_frag_nk(const __bf16* base, int ld) {
  const int lane = threadIdx.x & 31;
  const __bf16* p = base + (lane & 15) * ld + ((lane >> 4) << 3);
  return combine8(*(const bf16x8*)p, *(const bf16x8*)(p + 16));
}

static __device__ __forceinline__ floatx8 zero8() {
  floatx8 z;
#pragma unroll
  for (int e = 0; e < 8; ++e) z[e] = 0.0f;
  return z;
}

// Async VMEM->LDS copy of one 16B chunk (per-lane addresses), ASYNCcnt-tracked.
static __device__ __forceinline__ void async_copy_b128(const __bf16* gsrc,
                                                       __bf16* lds_dst) {
  const unsigned laddr = (unsigned)(size_t)(lds_bf16*)lds_dst;  // LDS byte addr
  asm volatile("global_load_async_to_lds_b128 %0, %1, off"
               :: "v"(laddr), "v"(gsrc) : "memory");
}

static __device__ __forceinline__ void wait_async() {
  asm volatile("s_wait_asynccnt 0" ::: "memory");
}

// ---------------------------------------------------------------------------
// Tiled bf16 WMMA GEMM:  C = epilogue( scale * A@B + bias ) [+ resid]
//   A: [M][K] bf16 row-major (lda);  B: [N][K] bf16 row-major (ldb) -- B is
//   always consumed in transposed storage so staging and fragment loads are
//   contiguous b128 accesses.
//   block = 256 threads = 8 waves; block tile 128x128, K-step 32;
//   wave (wm 0..3, wn 0..1) owns 32x64 = 2x4 WMMA tiles.
//   Tiles are staged with global_load_async_to_lds_b128 (TDM-path, ASYNCcnt).
// ---------------------------------------------------------------------------
template <bool HAS_BIAS, bool HAS_RES, bool DO_GELU, bool OUT_BF16>
__global__ void __launch_bounds__(256, 2)
gemm_bf16_kernel(const __bf16* __restrict__ A, const __bf16* __restrict__ B,
                 const float* __restrict__ bias, const float* __restrict__ resid,
                 void* __restrict__ Cout,
                 int N, int K, int lda, int ldb, int ldc,
                 long long strideA, long long strideB, long long strideC,
                 float scale) {
  __shared__ __bf16 As[128 * 40];   // [m][k], 32 cols + 8 pad
  __shared__ __bf16 Bs[128 * 40];   // [n][k], 32 cols + 8 pad

  const int tid  = threadIdx.x;
  const int lane = tid & 31;
  const int wid  = tid >> 5;
  const int wm   = wid >> 1;     // 0..3
  const int wn   = wid & 1;      // 0..1
  const int m0   = blockIdx.y * 128;
  const int n0   = blockIdx.x * 128;

  A += (long long)blockIdx.z * strideA;
  B += (long long)blockIdx.z * strideB;
  const long long cbase = (long long)blockIdx.z * strideC;

  floatx8 acc[2][4];
#pragma unroll
  for (int i = 0; i < 2; ++i)
#pragma unroll
    for (int j = 0; j < 4; ++j) acc[i][j] = zero8();

  for (int k0 = 0; k0 < K; k0 += 32) {
    // ---- stage A[128x32] and B[128x32] tiles into LDS ----
#pragma unroll
    for (int c = 0; c < 2; ++c) {
      const int chunk = tid + c * 256;        // 0..511, 8 bf16 each
      const int row = chunk >> 2;             // 0..127
      const int col = (chunk & 3) * 8;        // 0,8,16,24
      const __bf16* gA = A + (long long)(m0 + row) * lda + k0 + col;
      const __bf16* gB = B + (long long)(n0 + row) * ldb + k0 + col;
#if ASYNC_STAGE
      async_copy_b128(gA, &As[row * 40 + col]);
      async_copy_b128(gB, &Bs[row * 40 + col]);
#else
      *(bf16x8*)(&As[row * 40 + col]) = *(const bf16x8*)gA;
      *(bf16x8*)(&Bs[row * 40 + col]) = *(const bf16x8*)gB;
#endif
    }
    if (k0 + 32 < K) {   // hint next K-step into cache (global_prefetch_b8)
      const int row = tid >> 2, col = (tid & 3) * 8;
      __builtin_prefetch(A + (long long)(m0 + row) * lda + k0 + 32 + col, 0, 0);
      __builtin_prefetch(B + (long long)(n0 + row) * ldb + k0 + 32 + col, 0, 0);
    }
#if ASYNC_STAGE
    wait_async();          // our wave's async stores to LDS have landed
#endif
    __syncthreads();       // all waves' tiles visible

    bf16x16 af[2], bfr[4];
#pragma unroll
    for (int i = 0; i < 2; ++i)
      af[i] = load_a_frag(&As[(wm * 32 + i * 16) * 40], 40);
#pragma unroll
    for (int j = 0; j < 4; ++j)
      bfr[j] = load_b_frag_nk(&Bs[(wn * 64 + j * 16) * 40], 40);
#pragma unroll
    for (int i = 0; i < 2; ++i)
#pragma unroll
      for (int j = 0; j < 4; ++j) acc[i][j] = wmma_bf16(af[i], bfr[j], acc[i][j]);
    __syncthreads();
  }

  // ---- epilogue (C/D layout: elem e -> M = e + 8*(lane>=16), N = lane&15) ----
  const int hi = lane >> 4;
  const int nl = lane & 15;
#pragma unroll
  for (int i = 0; i < 2; ++i) {
#pragma unroll
    for (int j = 0; j < 4; ++j) {
#pragma unroll
      for (int e = 0; e < 8; ++e) {
        const int m = m0 + wm * 32 + i * 16 + e + hi * 8;
        const int n = n0 + wn * 64 + j * 16 + nl;
        float v = acc[i][j][e] * scale;
        if (HAS_BIAS) v += bias[n];
        if (DO_GELU) {
          const float x3 = v * v * v;
          v = 0.5f * v * (1.0f + tanhf(0.7978845608028654f * (v + 0.044715f * x3)));
        }
        const long long idx = cbase + (long long)m * ldc + n;
        if (HAS_RES) v += resid[idx];
        if (OUT_BF16) ((__bf16*)Cout)[idx] = (__bf16)v;
        else          ((float*)Cout)[idx] = v;
      }
    }
  }
  (void)N;
}

// ---------------------------------------------------------------------------
// Single-wave flash attention for the 16-head self-attention (hd = 64).
// One wave: 16 queries of one (b,h); loops over 32-key blocks.
// qkv: [B*S, 3D] bf16 (q|k|v).  vT: v transposed to [D][B*S] bf16.
// ---------------------------------------------------------------------------
__global__ void __launch_bounds__(32)
flash_mha_kernel(const __bf16* __restrict__ qkv, const __bf16* __restrict__ vT,
                 __bf16* __restrict__ out, int B, int S, int NH, int HD,
                 float scale) {
  __shared__ float  Sc[16 * 32];
  __shared__ __bf16 Pt[16 * 40];
  __shared__ float  alphaS[16];
  __shared__ float  lS[16];

  const int lane = threadIdx.x;
  const int nqb  = S / 16;
  const int qb   = blockIdx.x % nqb;
  const int h    = (blockIdx.x / nqb) % NH;
  const int b    = blockIdx.x / (nqb * NH);
  const int q0   = qb * 16;
  const int D    = NH * HD;
  const int D3   = 3 * D;
  const int BS   = B * S;

  const __bf16* qp    = qkv + (long long)(b * S + q0) * D3 + h * HD;
  const __bf16* kbase = qkv + (long long)b * S * D3 + D + h * HD;
  const __bf16* vtb   = vT + (long long)(h * HD) * BS + b * S;   // [d][key]

  const bf16x16 aq0 = load_a_frag(qp, D3);        // d = 0..31
  const bf16x16 aq1 = load_a_frag(qp + 32, D3);   // d = 32..63

  floatx8 O[4];
#pragma unroll
  for (int t = 0; t < 4; ++t) O[t] = zero8();
  float mstate = -3.4e38f, lstate = 0.0f;
  const int hi = lane >> 4, nl = lane & 15;

  for (int kb = 0; kb < S; kb += 32) {
    // scores S[16q x 32k] = Q @ K^T : K rows are [key][d] = [N][K] naturally
    floatx8 c0 = zero8(), c1 = zero8();
    {
      const __bf16* kp = kbase + (long long)kb * D3;
      bf16x16 bk;
      bk = load_b_frag_nk(kp, D3);                c0 = wmma_bf16(aq0, bk, c0);
      bk = load_b_frag_nk(kp + 32, D3);           c0 = wmma_bf16(aq1, bk, c0);
      bk = load_b_frag_nk(kp + 16 * D3, D3);      c1 = wmma_bf16(aq0, bk, c1);
      bk = load_b_frag_nk(kp + 16 * D3 + 32, D3); c1 = wmma_bf16(aq1, bk, c1);
    }
#pragma unroll
    for (int e = 0; e < 8; ++e) {
      Sc[(e + hi * 8) * 32 + nl]      = c0[e] * scale;
      Sc[(e + hi * 8) * 32 + 16 + nl] = c1[e] * scale;
    }
    __syncthreads();

    if (lane < 16) {   // online softmax per query row (no WMMA in this region)
      const int r = lane;
      float mx = mstate;
      for (int j = 0; j < 32; ++j) mx = fmaxf(mx, Sc[r * 32 + j]);
      const float alpha = __expf(mstate - mx);
      float s = 0.0f;
      for (int j = 0; j < 32; ++j) {
        const float p = __expf(Sc[r * 32 + j] - mx);
        s += p;
        Pt[r * 40 + j] = (__bf16)p;
      }
      lstate = lstate * alpha + s;
      mstate = mx;
      alphaS[r] = alpha;
    }
    __syncthreads();

    float arow[8];
#pragma unroll
    for (int e = 0; e < 8; ++e) arow[e] = alphaS[e + hi * 8];
#pragma unroll
    for (int t = 0; t < 4; ++t)
#pragma unroll
      for (int e = 0; e < 8; ++e) O[t][e] *= arow[e];

    // O += P @ V : V supplied transposed [d][key] -> contiguous b128 frags
    const bf16x16 ap = load_a_frag(Pt, 40);
#pragma unroll
    for (int t = 0; t < 4; ++t) {
      bf16x16 bv = load_b_frag_nk(vtb + (long long)(t * 16) * BS + kb, BS);
      O[t] = wmma_bf16(ap, bv, O[t]);
    }
    __syncthreads();
  }

  if (lane < 16) lS[lane] = lstate;
  __syncthreads();
  float inv[8];
#pragma unroll
  for (int e = 0; e < 8; ++e) inv[e] = 1.0f / lS[e + hi * 8];

  __bf16* op = out + (long long)(b * S + q0) * D + h * HD;
#pragma unroll
  for (int t = 0; t < 4; ++t)
#pragma unroll
    for (int e = 0; e < 8; ++e)
      op[(long long)(e + hi * 8) * D + t * 16 + nl] = (__bf16)(O[t][e] * inv[e]);
}

// ---------------------------------------------------------------------------
// Transpose + convert: in f32 [R][C] row-major -> out bf16 [C][R] row-major.
// LDS-tiled 32x32 so both global sides are coalesced.
// ---------------------------------------------------------------------------
__global__ void __launch_bounds__(256)
transpose_f32_bf16_kernel(const float* __restrict__ in, __bf16* __restrict__ out,
                          int R, int C) {
  __shared__ float t[32][33];
  const int r0 = blockIdx.y * 32, c0 = blockIdx.x * 32;
  const int tx = threadIdx.x & 31, ty = threadIdx.x >> 5;   // ty 0..7
#pragma unroll
  for (int i = 0; i < 32; i += 8)
    t[ty + i][tx] = in[(long long)(r0 + ty + i) * C + c0 + tx];
  __syncthreads();
#pragma unroll
  for (int i = 0; i < 32; i += 8)
    out[(long long)(c0 + ty + i) * R + r0 + tx] = (__bf16)t[tx][ty + i];
}

// bf16 sub-matrix transpose: out[c*ldout + r] = in[r*ldin + c]
__global__ void __launch_bounds__(256)
transpose_bf16_kernel(const __bf16* __restrict__ in, int ldin,
                      __bf16* __restrict__ out, int ldout) {
  __shared__ __bf16 t[32][34];
  const int r0 = blockIdx.y * 32, c0 = blockIdx.x * 32;
  const int tx = threadIdx.x & 31, ty = threadIdx.x >> 5;
#pragma unroll
  for (int i = 0; i < 32; i += 8)
    t[ty + i][tx] = in[(long long)(r0 + ty + i) * ldin + c0 + tx];
  __syncthreads();
#pragma unroll
  for (int i = 0; i < 32; i += 8)
    out[(long long)(c0 + ty + i) * ldout + r0 + tx] = t[tx][ty + i];
}

// ---------------------------------------------------------------------------
// LayerNorm over rows of length D, f32 in -> bf16 out
// ---------------------------------------------------------------------------
__global__ void __launch_bounds__(256)
layernorm_bf16_kernel(const float* __restrict__ x, const float* __restrict__ g,
                      const float* __restrict__ b, __bf16* __restrict__ y, int D) {
  __shared__ float red[256];
  const long long row = blockIdx.x;
  const float* xr = x + row * D;
  float s = 0.f, s2 = 0.f;
  for (int i = threadIdx.x; i < D; i += 256) { const float v = xr[i]; s += v; s2 += v * v; }
  red[threadIdx.x] = s; __syncthreads();
  for (int o = 128; o > 0; o >>= 1) { if (threadIdx.x < o) red[threadIdx.x] += red[threadIdx.x + o]; __syncthreads(); }
  const float mean = red[0] / D; __syncthreads();
  red[threadIdx.x] = s2; __syncthreads();
  for (int o = 128; o > 0; o >>= 1) { if (threadIdx.x < o) red[threadIdx.x] += red[threadIdx.x + o]; __syncthreads(); }
  const float var = red[0] / D - mean * mean;
  const float rstd = rsqrtf(var + 1e-5f);
  __bf16* yr = y + row * D;
  for (int i = threadIdx.x; i < D; i += 256)
    yr[i] = (__bf16)((xr[i] - mean) * rstd * g[i] + b[i]);
}

// ---------------------------------------------------------------------------
// Row softmax (scores pre-scaled), f32 in -> normalized bf16 out
// ---------------------------------------------------------------------------
__global__ void __launch_bounds__(256)
softmax_row_kernel(const float* __restrict__ S, __bf16* __restrict__ P, int Ncol) {
  __shared__ float red[256];
  const long long row = blockIdx.x;
  const float* sr = S + row * Ncol;
  float mx = -3.4e38f;
  for (int i = threadIdx.x; i < Ncol; i += 256) mx = fmaxf(mx, sr[i]);
  red[threadIdx.x] = mx; __syncthreads();
  for (int o = 128; o > 0; o >>= 1) { if (threadIdx.x < o) red[threadIdx.x] = fmaxf(red[threadIdx.x], red[threadIdx.x + o]); __syncthreads(); }
  mx = red[0]; __syncthreads();
  float sum = 0.f;
  for (int i = threadIdx.x; i < Ncol; i += 256) sum += __expf(sr[i] - mx);
  red[threadIdx.x] = sum; __syncthreads();
  for (int o = 128; o > 0; o >>= 1) { if (threadIdx.x < o) red[threadIdx.x] += red[threadIdx.x + o]; __syncthreads(); }
  const float inv = 1.0f / red[0];
  __bf16* pr = P + row * Ncol;
  for (int i = threadIdx.x; i < Ncol; i += 256) pr[i] = (__bf16)(__expf(sr[i] - mx) * inv);
}

__global__ void __launch_bounds__(256)
f32_to_bf16_kernel(const float* __restrict__ in, __bf16* __restrict__ out, long long n) {
  const long long i = (long long)blockIdx.x * 256 + threadIdx.x;
  if (i < n) out[i] = (__bf16)in[i];
}

// ---------------------------------------------------------------------------
// Host-side orchestration
// ---------------------------------------------------------------------------
extern "C" void kernel_launch(void* const* d_in, const int* in_sizes, int n_in,
                              void* d_out, int out_size, void* d_ws, size_t ws_size,
                              hipStream_t stream) {
  (void)in_sizes; (void)n_in; (void)out_size; (void)ws_size;
  const float* x    = (const float*)d_in[0];
  const float* z    = (const float*)d_in[1];
  const float* g1   = (const float*)d_in[2];
  const float* b1   = (const float*)d_in[3];
  const float* wqkv = (const float*)d_in[4];
  const float* bqkv = (const float*)d_in[5];
  const float* wo   = (const float*)d_in[6];
  const float* bo   = (const float*)d_in[7];
  const float* wkv  = (const float*)d_in[8];
  const float* bkv  = (const float*)d_in[9];
  const float* g2   = (const float*)d_in[10];
  const float* b2   = (const float*)d_in[11];
  const float* wm1  = (const float*)d_in[12];
  const float* bm1  = (const float*)d_in[13];
  const float* wm2  = (const float*)d_in[14];
  const float* bm2  = (const float*)d_in[15];
  float* outp = (float*)d_out;

  const int Bb = 4, S = 1024, D = 1024, NH = 16, HD = 64;
  const int BS = Bb * S;               // 4096
  const int D3 = 3 * D, D2 = 2 * D, H = 4 * D;

  char* ws = (char*)d_ws;
  size_t off = 0;
  auto alloc = [&](size_t bytes) -> char* {
    char* p = ws + off;
    off += (bytes + 255) & ~(size_t)255;
    return p;
  };

  __bf16* xn    = (__bf16*)alloc((size_t)BS * D * 2);
  __bf16* wqkvT = (__bf16*)alloc((size_t)D * D3 * 2);   // [3D][D]
  __bf16* qkvb  = (__bf16*)alloc((size_t)BS * D3 * 2);
  __bf16* vTs   = (__bf16*)alloc((size_t)BS * D * 2);   // self-attn v, [D][BS]
  __bf16* aob   = (__bf16*)alloc((size_t)BS * D * 2);
  __bf16* woT   = (__bf16*)alloc((size_t)D * D * 2);    // [D][D]
  float*  qf    = (float*)alloc((size_t)BS * D * 4);
  __bf16* qbf   = (__bf16*)alloc((size_t)BS * D * 2);
  __bf16* zbf   = (__bf16*)alloc((size_t)BS * D * 2);
  __bf16* wkvT  = (__bf16*)alloc((size_t)D * D2 * 2);   // [2D][D]
  __bf16* kvb   = (__bf16*)alloc((size_t)BS * D2 * 2);
  __bf16* vT    = (__bf16*)alloc((size_t)BS * D * 2);   // cross v, [D][BS]
  float*  scf   = (float*)alloc((size_t)Bb * S * S * 4);
  __bf16* pbf   = (__bf16*)alloc((size_t)Bb * S * S * 2);
  float*  outf  = (float*)alloc((size_t)BS * D * 4);
  __bf16* hnb   = (__bf16*)alloc((size_t)BS * D * 2);
  __bf16* wm1T  = (__bf16*)alloc((size_t)D * H * 2);    // [H][D]
  __bf16* h1b   = (__bf16*)alloc((size_t)BS * H * 2);
  __bf16* wm2T  = (__bf16*)alloc((size_t)H * D * 2);    // [D][H]

  const dim3 blk(256);

  // ---- one-time weight transposes (f32 [K][N] -> bf16 [N][K]) ----
  transpose_f32_bf16_kernel<<<dim3(D3 / 32, D / 32), blk, 0, stream>>>(wqkv, wqkvT, D, D3);
  transpose_f32_bf16_kernel<<<dim3(D  / 32, D / 32), blk, 0, stream>>>(wo,   woT,   D, D);
  transpose_f32_bf16_kernel<<<dim3(D2 / 32, D / 32), blk, 0, stream>>>(wkv,  wkvT,  D, D2);
  transpose_f32_bf16_kernel<<<dim3(H  / 32, D / 32), blk, 0, stream>>>(wm1,  wm1T,  D, H);
  transpose_f32_bf16_kernel<<<dim3(D  / 32, H / 32), blk, 0, stream>>>(wm2,  wm2T,  H, D);

  f32_to_bf16_kernel<<<dim3(BS * D / 256), blk, 0, stream>>>(z, zbf, (long long)BS * D);

  // ln1(x) -> xn
  layernorm_bf16_kernel<<<dim3(BS), blk, 0, stream>>>(x, g1, b1, xn, D);

  // qkv = xn @ w_qkv + b_qkv  [4096 x 3072] bf16
  gemm_bf16_kernel<true, false, false, true>
      <<<dim3(D3 / 128, BS / 128, 1), blk, 0, stream>>>(
          xn, wqkvT, bqkv, nullptr, qkvb, D3, D, D, D, D3, 0, 0, 0, 1.0f);

  // transpose self-attn v: qkv[:, 2D:3D] -> vTs [D][BS]
  transpose_bf16_kernel<<<dim3(D / 32, BS / 32), blk, 0, stream>>>(
      qkvb + 2 * D, D3, vTs, BS);

  // 16-head flash self-attention -> aob (bf16)
  flash_mha_kernel<<<dim3(Bb * NH * (S / 16)), dim3(32), 0, stream>>>(
      qkvb, vTs, aob, Bb, S, NH, HD, 0.125f);

  // qf = aob @ w_o + b_o + x   (f32)
  gemm_bf16_kernel<true, true, false, false>
      <<<dim3(D / 128, BS / 128, 1), blk, 0, stream>>>(
          aob, woT, bo, x, qf, D, D, D, D, D, 0, 0, 0, 1.0f);
  f32_to_bf16_kernel<<<dim3(BS * D / 256), blk, 0, stream>>>(qf, qbf, (long long)BS * D);

  // kv = z @ w_kv + b_kv  [4096 x 2048] bf16 (k = cols 0..D-1, v = D..2D-1)
  gemm_bf16_kernel<true, false, false, true>
      <<<dim3(D2 / 128, BS / 128, 1), blk, 0, stream>>>(
          zbf, wkvT, bkv, nullptr, kvb, D2, D, D, D, D2, 0, 0, 0, 1.0f);

  // transpose cross v: kv[:, D:2D] -> vT [D][BS]
  transpose_bf16_kernel<<<dim3(D / 32, BS / 32), blk, 0, stream>>>(
      kvb + D, D2, vT, BS);

  // scores = (q @ k^T) / 32  batched; k rows [key][d] are already [N][K]
  gemm_bf16_kernel<false, false, false, false>
      <<<dim3(S / 128, S / 128, Bb), blk, 0, stream>>>(
          qbf, kvb, nullptr, nullptr, scf, S, D, D, D2, S,
          (long long)S * D, (long long)S * D2, (long long)S * S, 1.0f / 32.0f);

  // row softmax -> P (bf16, normalized)
  softmax_row_kernel<<<dim3(Bb * S), blk, 0, stream>>>(scf, pbf, S);

  // outf = P @ v  batched; vT is [D][BS], per-batch key offset = b*S
  gemm_bf16_kernel<false, false, false, false>
      <<<dim3(D / 128, S / 128, Bb), blk, 0, stream>>>(
          pbf, vT, nullptr, nullptr, outf, D, S, S, BS, D,
          (long long)S * S, (long long)S, (long long)S * D, 1.0f);

  // ln2(outf) -> hnb
  layernorm_bf16_kernel<<<dim3(BS), blk, 0, stream>>>(outf, g2, b2, hnb, D);

  // h1 = gelu(hnb @ w_m1 + b_m1)  [4096 x 4096] bf16
  gemm_bf16_kernel<true, false, true, true>
      <<<dim3(H / 128, BS / 128, 1), blk, 0, stream>>>(
          hnb, wm1T, bm1, nullptr, h1b, H, D, D, D, H, 0, 0, 0, 1.0f);

  // out = h1 @ w_m2 + b_m2 + outf  (final, f32)
  gemm_bf16_kernel<true, true, false, false>
      <<<dim3(8, 32, 1), blk, 0, stream>>>(
          h1b, wm2T, bm2, outf, outp, D, H, H, H, D, 0, 0, 0, 1.0f);
}
